// GCN_2241972929175
// MI455X (gfx1250) — matrix operand order repaired and checked by reading, not compiled, not run on hardware
//
#include <hip/hip_runtime.h>

typedef __attribute__((ext_vector_type(16))) __bf16 v16bf;
typedef __attribute__((ext_vector_type(8)))  __bf16 v8bf;
typedef __attribute__((ext_vector_type(8)))  float  v8f;

static __device__ __forceinline__ __bf16 f2bf(float f) {
  unsigned u = __builtin_bit_cast(unsigned, f);
  u += 0x7FFFu + ((u >> 16) & 1u);          // round-to-nearest-even
  unsigned short s = (unsigned short)(u >> 16);
  return __builtin_bit_cast(__bf16, s);
}

// ---------------- elementwise helpers ----------------
__global__ void fill_f32(float* __restrict__ p, int n, float v) {
  int i = blockIdx.x * blockDim.x + threadIdx.x;
  if (i < n) p[i] = v;
}

__global__ void cvt_f32_bf16(const float* __restrict__ in, __bf16* __restrict__ out, int n) {
  int i = blockIdx.x * blockDim.x + threadIdx.x;
  if (i < n) out[i] = f2bf(in[i]);
}

// degree accumulation: dinv pre-filled with 1.0 (self loop)
__global__ void deg_edges(float* __restrict__ dinv, const int* __restrict__ dst, int E) {
  int e = blockIdx.x * blockDim.x + threadIdx.x;
  if (e < E) atomicAdd(&dinv[dst[e]], 1.0f);
}

__global__ void deg_to_dinv(float* __restrict__ dinv, int n) {
  int i = blockIdx.x * blockDim.x + threadIdx.x;
  if (i < n) { float d = dinv[i]; dinv[i] = (d > 0.0f) ? rsqrtf(d) : 0.0f; }
}

__global__ void norm_edges(const float* __restrict__ dinv, const int* __restrict__ src,
                           const int* __restrict__ dst, float* __restrict__ norm, int E) {
  int e = blockIdx.x * blockDim.x + threadIdx.x;
  if (e < E) norm[e] = dinv[src[e]] * dinv[dst[e]];
}

// ---------------- weight swizzle for WMMA B-fragments ----------------
// W: [K x C] f32 row-major -> Ws bf16 laid out as [nt][kt][lane][e] (e=0..15)
// per-lane B fragment: col = nt*16 + (lane&15), K = kt*32 + (e<8 ? 8*(lane>>4) + e : 16 + 8*(lane>>4) + e-8)
__global__ void swizzle_w(const float* __restrict__ W, __bf16* __restrict__ Ws, int K, int C) {
  int idx = blockIdx.x * blockDim.x + threadIdx.x;
  if (idx >= K * C) return;
  int e    = idx & 15;
  int lane = (idx >> 4) & 31;
  int kt   = (idx >> 9) % (K >> 5);
  int nt   = (idx >> 9) / (K >> 5);
  int half = lane >> 4, l = lane & 15;
  int k = kt * 32 + (e < 8 ? 8 * half + e : 16 + 8 * half + (e - 8));
  int n = nt * 16 + l;
  Ws[idx] = f2bf(W[k * C + n]);
}

// ---------------- GEMM: T[M x C] = A[M x K](bf16) @ Ws (swizzled W, bf16), f32 acc ----
// block (32,4): 4 waves, each wave owns one 16-col tile; grid (M/16, C/64)
__global__ void gemm_bf16_wmma(const __bf16* __restrict__ A, const __bf16* __restrict__ Ws,
                               float* __restrict__ T, int M, int K, int C) {
  int lane = threadIdx.x;
  int nt   = blockIdx.y * 4 + threadIdx.y;     // 16-wide column tile
  int n0   = nt * 16;
  int m0   = blockIdx.x * 16;
  int half = lane >> 4, l = lane & 15;
  int mr   = m0 + l; if (mr > M - 1) mr = M - 1;
  const __bf16* Arow  = A + (size_t)mr * K;
  const __bf16* Bbase = Ws + ((size_t)nt * (K >> 5)) * 512 + lane * 16;

  v8f acc = {};
  for (int k0 = 0; k0 < K; k0 += 32) {
    v8bf alo = *(const v8bf*)(Arow + k0 + 8 * half);
    v8bf ahi = *(const v8bf*)(Arow + k0 + 16 + 8 * half);
    v16bf a;
#pragma unroll
    for (int j = 0; j < 8; ++j) { a[j] = alo[j]; a[j + 8] = ahi[j]; }
    v16bf b = *(const v16bf*)(Bbase + (size_t)(k0 >> 5) * 512);
    acc = __builtin_amdgcn_wmma_f32_16x16x32_bf16(false, a, false, b, (short)0, acc,
                                                  false, false);
  }
#pragma unroll
  for (int r = 0; r < 8; ++r) {
    int row = m0 + 8 * half + r;
    if (row < M) T[(size_t)row * C + n0 + l] = acc[r];
  }
}

// ---------------- acc init: bias + self-loop contribution ----------------
__global__ void init_acc(const float* __restrict__ t, const float* __restrict__ dinv,
                         const float* __restrict__ bias, float* __restrict__ acc,
                         int total, int cShift, int cMask) {
  int idx = blockIdx.x * blockDim.x + threadIdx.x;
  if (idx >= total) return;
  int i = idx >> cShift, c = idx & cMask;
  float dn = dinv[i];
  acc[idx] = bias[c] + t[idx] * dn * dn;
}

// ---------------- edge scatter: acc[dst] += t[src] * norm ----------------
// blockDim = (C/4, 256/(C/4)); one float4 channel-group per thread.x
__global__ void scatter_edges(const float4* __restrict__ t4, float* __restrict__ acc,
                              const int* __restrict__ src, const int* __restrict__ dst,
                              const float* __restrict__ norm, int E, int C4, int C) {
  int e = blockIdx.x * blockDim.y + threadIdx.y;
  if (e >= E) return;
  int s = src[e], d = dst[e];
  float w = norm[e];
  int g = threadIdx.x;
  float4 v = t4[(size_t)s * C4 + g];
  float* out = acc + (size_t)d * C + g * 4;
  atomicAdd(out + 0, v.x * w);
  atomicAdd(out + 1, v.y * w);
  atomicAdd(out + 2, v.z * w);
  atomicAdd(out + 3, v.w * w);
}

// ---------------- ReLU + bf16 convert (next layer's A) ----------------
__global__ void relu_bf16(const float* __restrict__ acc, __bf16* __restrict__ h, int n) {
  int i = blockIdx.x * blockDim.x + threadIdx.x;
  if (i < n) h[i] = f2bf(fmaxf(acc[i], 0.0f));
}

// ---------------- mean pool (scatter phase) ----------------
__global__ void pool_scatter(const float4* __restrict__ h4, const int* __restrict__ batch,
                             float* __restrict__ pooled, float* __restrict__ cnt, int Nn) {
  int i = blockIdx.x * blockDim.y + threadIdx.y;
  if (i >= Nn) return;
  int b = batch[i];
  int g = threadIdx.x;                 // 0..15 (C=64)
  float4 v = h4[(size_t)i * 16 + g];
  float* out = pooled + (size_t)b * 64 + g * 4;
  atomicAdd(out + 0, v.x);
  atomicAdd(out + 1, v.y);
  atomicAdd(out + 2, v.z);
  atomicAdd(out + 3, v.w);
  if (g == 0) atomicAdd(&cnt[b], 1.0f);
}

// ---------------- head: mean, linear 64->10, log_softmax + softmax -------
__global__ void head_kernel(const float* __restrict__ pooled, const float* __restrict__ cnt,
                            const float* __restrict__ Wlin, const float* __restrict__ blin,
                            float* __restrict__ out, int G) {
  int g = blockIdx.x * blockDim.x + threadIdx.x;
  if (g >= G) return;
  float inv = 1.0f / fmaxf(cnt[g], 1.0f);
  float logit[10];
#pragma unroll
  for (int c = 0; c < 10; ++c) logit[c] = blin[c];
  for (int k = 0; k < 64; ++k) {
    float p = pooled[g * 64 + k] * inv;
#pragma unroll
    for (int c = 0; c < 10; ++c) logit[c] = fmaf(p, Wlin[k * 10 + c], logit[c]);
  }
  float mx = logit[0];
#pragma unroll
  for (int c = 1; c < 10; ++c) mx = fmaxf(mx, logit[c]);
  float ex[10], se = 0.0f;
#pragma unroll
  for (int c = 0; c < 10; ++c) { ex[c] = expf(logit[c] - mx); se += ex[c]; }
  float lse = logf(se);
#pragma unroll
  for (int c = 0; c < 10; ++c) {
    out[g * 10 + c]          = (logit[c] - mx) - lse;   // log_softmax
    out[G * 10 + g * 10 + c] = ex[c] / se;              // softmax
  }
}

extern "C" void kernel_launch(void* const* d_in, const int* in_sizes, int n_in,
                              void* d_out, int out_size, void* d_ws, size_t ws_size,
                              hipStream_t stream) {
  (void)n_in; (void)ws_size;
  const float* x     = (const float*)d_in[0];
  const int*   ei    = (const int*)d_in[1];
  const int*   batch = (const int*)d_in[2];
  const float* W[6]  = {(const float*)d_in[4],  (const float*)d_in[6],  (const float*)d_in[8],
                        (const float*)d_in[10], (const float*)d_in[12], (const float*)d_in[14]};
  const float* bia[6]= {(const float*)d_in[5],  (const float*)d_in[7],  (const float*)d_in[9],
                        (const float*)d_in[11], (const float*)d_in[13], (const float*)d_in[15]};
  const float* Wlin = (const float*)d_in[16];
  const float* blin = (const float*)d_in[17];

  const int Nn = in_sizes[2];              // 100000 nodes
  const int E  = in_sizes[1] / 2;          // 3.2M edges
  const int H  = in_sizes[5];              // 256
  const int F  = in_sizes[0] / Nn;         // 128
  const int G  = out_size / 20;            // 512 graphs
  const int* src = ei;
  const int* dst = ei + E;

  int Ks[6] = {F, H, H, H,     H / 2, H / 2};
  int Cs[6] = {H, H, H, H / 2, H / 2, H / 4};

  // ---- workspace carve-out (256B aligned) ----
  size_t off = 0;
  auto carve = [&](size_t bytes) -> void* {
    void* p = (char*)d_ws + off;
    off += (bytes + 255) & ~(size_t)255;
    return p;
  };
  float*  dinv   = (float*)carve((size_t)Nn * 4);
  float*  norm   = (float*)carve((size_t)E * 4);
  float*  tbuf   = (float*)carve((size_t)Nn * H * 4);
  float*  accbuf = (float*)carve((size_t)Nn * H * 4);
  __bf16* hbf    = (__bf16*)carve((size_t)Nn * H * 2);
  __bf16* Ws[6];
  for (int i = 0; i < 6; ++i) Ws[i] = (__bf16*)carve((size_t)Ks[i] * Cs[i] * 2);
  float*  pooled = (float*)carve((size_t)G * 64 * 4);
  float*  cnt    = (float*)carve((size_t)G * 4);

  const int TB = 256;
  // ---- degree / dinv / edge norms ----
  fill_f32<<<(Nn + TB - 1) / TB, TB, 0, stream>>>(dinv, Nn, 1.0f);   // self loop
  deg_edges<<<(E + TB - 1) / TB, TB, 0, stream>>>(dinv, dst, E);
  deg_to_dinv<<<(Nn + TB - 1) / TB, TB, 0, stream>>>(dinv, Nn);
  norm_edges<<<(E + TB - 1) / TB, TB, 0, stream>>>(dinv, src, dst, norm, E);

  // ---- weight swizzle + input convert ----
  for (int i = 0; i < 6; ++i) {
    int n = Ks[i] * Cs[i];
    swizzle_w<<<(n + TB - 1) / TB, TB, 0, stream>>>(W[i], Ws[i], Ks[i], Cs[i]);
  }
  cvt_f32_bf16<<<((Nn * F) + TB - 1) / TB, TB, 0, stream>>>(x, hbf, Nn * F);

  // ---- 6 GCN layers ----
  for (int L = 0; L < 6; ++L) {
    int K = Ks[L], C = Cs[L];
    dim3 gb((Nn + 15) / 16, C / 64 > 0 ? C / 64 : 1);
    if (C < 64) gb.y = 1;
    gemm_bf16_wmma<<<gb, dim3(32, 4), 0, stream>>>(hbf, Ws[L], tbuf, Nn, K, C);

    int total = Nn * C;
    int cShift = __builtin_ctz(C);
    init_acc<<<(total + TB - 1) / TB, TB, 0, stream>>>(tbuf, dinv, bia[L], accbuf,
                                                       total, cShift, C - 1);

    int C4 = C / 4;
    int epb = 256 / C4;
    scatter_edges<<<(E + epb - 1) / epb, dim3(C4, epb), 0, stream>>>(
        (const float4*)tbuf, accbuf, src, dst, norm, E, C4, C);

    if (L < 5)
      relu_bf16<<<(total + TB - 1) / TB, TB, 0, stream>>>(accbuf, hbf, total);
  }

  // ---- pooling + head ----
  fill_f32<<<(G * 64 + TB - 1) / TB, TB, 0, stream>>>(pooled, G * 64, 0.0f);
  fill_f32<<<(G + TB - 1) / TB, TB, 0, stream>>>(cnt, G, 0.0f);
  pool_scatter<<<(Nn + 15) / 16, dim3(16, 16), 0, stream>>>(
      (const float4*)accbuf, batch, pooled, cnt, Nn);
  head_kernel<<<(G + TB - 1) / TB, TB, 0, stream>>>(pooled, cnt, Wlin, blin,
                                                    (float*)d_out, G);
}
